// PixelWiseSupervisedContrastiveLoss_41068477284615
// MI455X (gfx1250) — matrix operand order repaired and checked by reading, not compiled, not run on hardware
//
#include <hip/hip_runtime.h>
#include <hip/hip_bf16.h>

// ---------------- problem constants ----------------
#define T_REAL   14884          // 4*61*61 pixels
#define C_DIM    256
#define HWP      3721           // 61*61
#define T_PAD    14976          // multiple of 128 (rows/block) and 16 (tile)
#define N_COL_TILES (T_PAD / 16)        // 936
#define N_SLICES 8
#define TILES_PER_SLICE (N_COL_TILES / N_SLICES)  // 117
#define ROWS_PER_BLOCK 128
#define N_ROW_BLOCKS (T_PAD / ROWS_PER_BLOCK)     // 117
#define INV_TEMP (1.0f / 0.07f)
#define LDS_STRIDE 264          // 256 + 8 halves padding -> conflict-free 64-bank access

// workspace layout (bytes)
#define FEATS_BYTES ((size_t)T_PAD * C_DIM * 2)   // fp16 normalized feats
#define LAB_BYTES   ((size_t)T_PAD * 4)           // int32 labels

typedef __attribute__((ext_vector_type(16))) _Float16 v16h;
typedef __attribute__((ext_vector_type(8)))  float    v8f;

union Frag16 { v16h h; uint4 q[2]; };

// ---------------------------------------------------------------------------
// Pass 1: L2-normalize each pixel vector, emit fp16 feats (row-major T_PAD x 256)
// and int32 labels. Coalesced: lanes sweep pixels, waves sweep channel groups.
// ---------------------------------------------------------------------------
__global__ __launch_bounds__(256)
void scl_normalize(const float* __restrict__ emb,
                   const long long* __restrict__ tgt,
                   _Float16* __restrict__ feats,
                   int* __restrict__ labels) {
    __shared__ float ssq[32];
    const int tid  = threadIdx.x;
    const int lane = tid & 31;          // pixel within block's group of 32
    const int w    = tid >> 5;          // channel group (32 channels each)
    const int t    = blockIdx.x * 32 + lane;
    const bool valid = (t < T_REAL);
    const int n = t / HWP;
    const int p = t - n * HWP;
    const float* src = emb + (size_t)n * (C_DIM * HWP) + (size_t)(w * 32) * HWP + p;

    float v[32];
    float s = 0.f;
#pragma unroll
    for (int i = 0; i < 32; ++i) {
        float x = valid ? src[(size_t)i * HWP] : 0.f;
        v[i] = x;
        s += x * x;
    }
    if (tid < 32) ssq[tid] = 0.f;
    __syncthreads();
    atomicAdd(&ssq[lane], s);           // ds_add_f32
    __syncthreads();

    const float nrm   = fmaxf(sqrtf(ssq[lane]), 1e-12f);
    const float scale = 1.0f / nrm;

    union { _Float16 h[32]; uint4 q[4]; } outp;
#pragma unroll
    for (int i = 0; i < 32; ++i) outp.h[i] = (_Float16)(v[i] * scale);
    uint4* dst = (uint4*)(feats + (size_t)t * C_DIM + w * 32);
#pragma unroll
    for (int i = 0; i < 4; ++i) dst[i] = outp.q[i];

    if (tid < 32) labels[t] = valid ? (int)tgt[t] : -1;
}

// ---------------------------------------------------------------------------
// Pass 2: fused feats @ feats^T -> exp -> masked row sums.
// Each wave owns 16 rows (A register-resident, 8 x v16h).
// 16-col B tiles streamed into double-buffered LDS with
// GLOBAL_LOAD_ASYNC_TO_LDS_B128 (ASYNCcnt), shared by the 8 waves.
// Per-slice partial sums written (deterministic, no atomics).
// ---------------------------------------------------------------------------
__global__ __launch_bounds__(256)
void scl_gemm_exp(const _Float16* __restrict__ feats,
                  const int* __restrict__ labels,
                  float* __restrict__ pos_part,
                  float* __restrict__ all_part) {
    __shared__ _Float16 ldsB[2][16 * LDS_STRIDE];
    __shared__ int      ldsLab[2][16];

    const int tid   = threadIdx.x;
    const int lane  = tid & 31;
    const int wave  = tid >> 5;
    const int rbase = blockIdx.x * ROWS_PER_BLOCK + wave * 16;
    const int slice = blockIdx.y;
    const int j0    = slice * TILES_PER_SLICE;

    // SGPR base for GVS async addressing
    const unsigned long long featsBase = (unsigned long long)(uintptr_t)feats;

    // ---- A fragments: documented 16-bit A layout, 16x32 per fragment.
    // lane<16: row M=lane, halves K = kb+{0..7, 16..23}
    // lane>=16: row M=lane-16, halves K = kb+{8..15, 24..31}
    Frag16 a[8];
    {
        const int m   = lane & 15;
        const int sel = (lane < 16) ? 0 : 8;
        const _Float16* rp = feats + (size_t)(rbase + m) * C_DIM;
#pragma unroll
        for (int f = 0; f < 8; ++f) {
            const int kb = f * 32;
            a[f].q[0] = *(const uint4*)(rp + kb + sel);
            a[f].q[1] = *(const uint4*)(rp + kb + 16 + sel);
        }
    }

    // row labels for this lane's 8 output rows (C/D layout: lanes>=16 -> M+8)
    const int rowHalf = rbase + ((lane >> 4) << 3);
    int labRow[8];
#pragma unroll
    for (int g = 0; g < 8; ++g) labRow[g] = labels[rowHalf + g];

    float accAll[8], accPos[8];
#pragma unroll
    for (int g = 0; g < 8; ++g) { accAll[g] = 0.f; accPos[g] = 0.f; }

    // cooperative B-tile staging: 256 threads x 32B = 8KB tile, async to LDS.
    // IOFFSET advances both global and LDS addresses -> offset:0/16 copies 32B.
    auto stage = [&](int j, int buf) {
        const int col = tid >> 4;          // 0..15
        const int kp  = (tid & 15) * 16;   // half offset 0..240
        const unsigned int goff = (unsigned int)(((j * 16 + col) * C_DIM + kp) * 2);
        _Float16* dst = &ldsB[buf][col * LDS_STRIDE + kp];
        const unsigned int loff = (unsigned int)(uintptr_t)dst;  // addr[31:0] = LDS offset
        asm volatile(
            "global_load_async_to_lds_b128 %0, %1, %2 offset:0\n\t"
            "global_load_async_to_lds_b128 %0, %1, %2 offset:16"
            :
            : "v"(loff), "v"(goff), "s"(featsBase)
            : "memory");
        if (tid < 16) ldsLab[buf][tid] = labels[j * 16 + tid];
        // hint the tile after the staged one into cache (global_prefetch_b8)
        if (j + 1 < j0 + TILES_PER_SLICE)
            __builtin_prefetch(feats + (size_t)((j + 1) * 16 + col) * C_DIM + kp, 0, 1);
    };

    stage(j0, 0);
    asm volatile("s_wait_asynccnt 0" ::: "memory");
    __syncthreads();

    for (int jj = 0; jj < TILES_PER_SLICE; ++jj) {
        const int j   = j0 + jj;
        const int buf = jj & 1;
        if (jj + 1 < TILES_PER_SLICE) stage(j + 1, buf ^ 1);

        const int colIdx = j * 16 + (lane & 15);
        const int labCol = ldsLab[buf][lane & 15];

        // ---- 16x16 tile GEMM over K=256 : 8 chained WMMAs
        v8f c = {};
        {
            const int khalf = (lane < 16) ? 0 : 8;    // B layout: lanes 0-15 K {0..7,16..23}, 16-31 K {8..15,24..31}
            const _Float16* bp = &ldsB[buf][(lane & 15) * LDS_STRIDE + khalf];
#pragma unroll
            for (int f = 0; f < 8; ++f) {
                Frag16 b;
                b.q[0] = *(const uint4*)(bp + f * 32);
                b.q[1] = *(const uint4*)(bp + f * 32 + 16);
                c = __builtin_amdgcn_wmma_f32_16x16x32_f16(
                        /*neg_a=*/false, a[f].h, /*neg_b=*/false, b.h,
                        /*c_mod=*/(short)0, c, /*reuse_a=*/false, /*reuse_b=*/false);
            }
        }

        // ---- epilogue: exp + masks + accumulate
#pragma unroll
        for (int g = 0; g < 8; ++g) {
            const float e   = __expf(c[g] * INV_TEMP);
            const int   row = rowHalf + g;
            const bool  cv  = (colIdx < T_REAL) & (colIdx != row);
            accAll[g] += cv ? e : 0.f;
            accPos[g] += (cv & (labCol == labRow[g])) ? e : 0.f;
        }

        // my async writes to the other buffer must be LDS-visible before the
        // block-wide barrier releases readers of that buffer next iteration
        asm volatile("s_wait_asynccnt 0" ::: "memory");
        __syncthreads();
    }

    // ---- reduce across the 16 lanes holding one row (wave32 halves)
#pragma unroll
    for (int g = 0; g < 8; ++g) {
        float sa = accAll[g], sp = accPos[g];
#pragma unroll
        for (int m = 8; m >= 1; m >>= 1) {
            sa += __shfl_xor(sa, m, 16);
            sp += __shfl_xor(sp, m, 16);
        }
        if ((lane & 15) == 0) {
            const int row = rowHalf + g;
            if (row < T_REAL) {
                pos_part[(size_t)slice * T_PAD + row] = sp;
                all_part[(size_t)slice * T_PAD + row] = sa;
            }
        }
    }
}

// ---------------------------------------------------------------------------
// Pass 3: combine slice partials, per-row loss, mean over valid rows.
// ---------------------------------------------------------------------------
__global__ __launch_bounds__(256)
void scl_finalize(const float* __restrict__ pos_part,
                  const float* __restrict__ all_part,
                  float* __restrict__ out) {
    __shared__ float sLoss[256];
    __shared__ int   sCnt[256];
    const int tid = threadIdx.x;
    float loss = 0.f;
    int   cnt  = 0;
    for (int t = tid; t < T_REAL; t += 256) {
        float sp = 0.f, sa = 0.f;
#pragma unroll
        for (int s = 0; s < N_SLICES; ++s) {
            sp += pos_part[(size_t)s * T_PAD + t];
            sa += all_part[(size_t)s * T_PAD + t];
        }
        if (sp > 0.f) {
            loss += -(__logf(sp) - __logf(sa));
            cnt  += 1;
        }
    }
    sLoss[tid] = loss;
    sCnt[tid]  = cnt;
    __syncthreads();
    for (int o = 128; o > 0; o >>= 1) {
        if (tid < o) { sLoss[tid] += sLoss[tid + o]; sCnt[tid] += sCnt[tid + o]; }
        __syncthreads();
    }
    if (tid == 0) out[0] = sLoss[0] / (float)max(sCnt[0], 1);
}

// ---------------------------------------------------------------------------
extern "C" void kernel_launch(void* const* d_in, const int* in_sizes, int n_in,
                              void* d_out, int out_size, void* d_ws, size_t ws_size,
                              hipStream_t stream) {
    (void)in_sizes; (void)n_in; (void)out_size; (void)ws_size;
    const float*     emb = (const float*)d_in[0];
    const long long* tgt = (const long long*)d_in[1];

    char* ws = (char*)d_ws;
    _Float16* feats    = (_Float16*)ws;
    int*      labels   = (int*)(ws + FEATS_BYTES);
    float*    pos_part = (float*)(ws + FEATS_BYTES + LAB_BYTES);
    float*    all_part = pos_part + (size_t)N_SLICES * T_PAD;
    float*    out      = (float*)d_out;

    scl_normalize<<<T_PAD / 32, 256, 0, stream>>>(emb, tgt, feats, labels);

    dim3 g2(N_ROW_BLOCKS, N_SLICES);
    scl_gemm_exp<<<g2, 256, 0, stream>>>(feats, labels, pos_part, all_part);

    scl_finalize<<<1, 256, 0, stream>>>(pos_part, all_part, out);
}
